// MambaLayer_18476949307797
// MI455X (gfx1250) — matrix operand order repaired and checked by reading, not compiled, not run on hardware
//
#include <hip/hip_runtime.h>

typedef __attribute__((ext_vector_type(16))) _Float16 v16h;
typedef __attribute__((ext_vector_type(8)))  float    v8f;

#define L_SEQ   32768
#define NCH     96
#define DINNER  192
#define DSTATE  16
#define DTRANK  6
#define LDXZ    384
#define LDDBC   48
#define CHUNK   128
#define NCHUNK  256
#define MTILES  (L_SEQ / 16)   // 2048

// ---------------------------------------------------------------------------
// A-fragment mapping for v_wmma_f32_16x16x32_f16 (ISA 7.12.2, 16-bit A 16x32):
//   lane = M + 16*hs ; VGPR j, half hh hold K:
//   K 0..7  -> hs0 j=k/2        | K 8..15  -> hs1 j=(k-8)/2
//   K 16..23-> hs0 j=4+(k-16)/2 | K 24..31 -> hs1 j=4+(k-24)/2
__device__ __forceinline__ void a_map(int k, int& hs, int& j, int& hh) {
  hs = (k >> 3) & 1;
  j  = ((k & 7) >> 1) + ((k >> 4) << 2);
  hh = k & 1;
}

// ---------------------------------------------------------------------------
// Deterministic shell-order permutation: lexsort by (k, d, h, w).
// Shell k offset = 32768 - (32-2k)^3 (inner-cube closed form).
__global__ void build_perm(int* __restrict__ perm) {
  int k = threadIdx.x;
  if (k >= 16) return;
  int side = 32 - 2 * k;
  int pos  = 32768 - side * side * side;
  for (int d = 0; d < 32; ++d)
    for (int h = 0; h < 32; ++h)
      for (int w = 0; w < 32; ++w) {
        int m = min(min(min(d, 31 - d), min(h, 31 - h)), min(w, 31 - w));
        if (m == k) perm[pos++] = (d << 10) | (h << 5) | w;
      }
}

__device__ __forceinline__ int vox_map(int v, int axis) {
  int d = v >> 10, h = (v >> 5) & 31, w = v & 31;
  if (axis == 0) return (d << 10) | (h << 5) | w;
  if (axis == 1) return (h << 10) | (d << 5) | w;
  return (w << 10) | (h << 5) | d;
}

// ---------------------------------------------------------------------------
__global__ void init_out(const float* __restrict__ x, float* __restrict__ out) {
  int i = blockIdx.x * blockDim.x + threadIdx.x;
  if (i < NCH * L_SEQ) out[i] = x[i];
}

__global__ void neg_exp(const float* __restrict__ A_log, float* __restrict__ Aneg) {
  int i = blockIdx.x * blockDim.x + threadIdx.x;
  if (i < DINNER * DSTATE) Aneg[i] = -__expf(A_log[i]);
}

// ---------------------------------------------------------------------------
// Pack row-major f32 W[N][K] into B-fragments (32x16 f16 tiles per K-chunk).
// B layout: lane holds column N=lane%16; hs=lane/16; VGPR j half hh -> K = 2j+hh+8*hs (j<4), 2j+hh+8+8*hs (j>=4).
__global__ void pack_b(const float* __restrict__ W, _Float16* __restrict__ dst,
                       int Kdim, int KC, int Nrows, int Ntiles) {
  int t = blockIdx.x * blockDim.x + threadIdx.x;
  int total = Ntiles * KC * 32;
  if (t >= total) return;
  int lane = t & 31;
  int kc   = (t >> 5) % KC;
  int nt   = t / (32 * KC);
  int N    = nt * 16 + (lane & 15);
  int hs   = lane >> 4;
  _Float16* o = dst + (size_t)t * 16;
#pragma unroll
  for (int j = 0; j < 8; ++j)
#pragma unroll
    for (int hh = 0; hh < 2; ++hh) {
      int K = kc * 32 + ((j < 4) ? (2 * j + hh + 8 * hs) : (16 + 2 * (j - 4) + hh + 8 * hs));
      float v = (N < Nrows) ? W[(size_t)N * Kdim + K] : 0.f;
      o[j * 2 + hh] = (_Float16)v;
    }
}

// Pack row-major f32 src[L][Kdim] into A-fragments.
__global__ void pack_a(const float* __restrict__ src, _Float16* __restrict__ dst,
                       int Kdim, int KC) {
  int t = blockIdx.x * blockDim.x + threadIdx.x;
  int total = MTILES * KC * 32;
  if (t >= total) return;
  int lane = t & 31;
  int kc   = (t >> 5) % KC;
  int tile = t / (32 * KC);
  int M = lane & 15, hs = lane >> 4;
  int row = tile * 16 + M;
  _Float16* o = dst + (size_t)t * 16;
#pragma unroll
  for (int j = 0; j < 8; ++j)
#pragma unroll
    for (int hh = 0; hh < 2; ++hh) {
      int k = (j < 4) ? (2 * j + hh + 8 * hs) : (16 + 2 * (j - 4) + hh + 8 * hs);
      o[j * 2 + hh] = (_Float16)src[(size_t)row * Kdim + kc * 32 + k];
    }
}

// ---------------------------------------------------------------------------
// Gather voxels in shell order, LayerNorm over 96 channels (one wave per token,
// 3 channels per lane, wave32 butterfly reduce), emit A-fragments directly.
__global__ void gather_ln_pack(const float* __restrict__ x, const int* __restrict__ perm,
                               const float* __restrict__ ln_g, const float* __restrict__ ln_b,
                               _Float16* __restrict__ Afrag, int axis) {
  int l = blockIdx.x * 8 + (threadIdx.x >> 5);
  if (l >= L_SEQ) return;
  int lane = threadIdx.x & 31;
  int vox  = vox_map(perm[l], axis);

  float c0 = x[(size_t)(lane)      * L_SEQ + vox];
  float c1 = x[(size_t)(lane + 32) * L_SEQ + vox];
  float c2 = x[(size_t)(lane + 64) * L_SEQ + vox];

  float s = c0 + c1 + c2;
#pragma unroll
  for (int off = 16; off; off >>= 1) s += __shfl_xor(s, off, 32);
  float mu = s * (1.f / 96.f);
  float d0 = c0 - mu, d1 = c1 - mu, d2 = c2 - mu;
  float vv = d0 * d0 + d1 * d1 + d2 * d2;
#pragma unroll
  for (int off = 16; off; off >>= 1) vv += __shfl_xor(vv, off, 32);
  float rs = rsqrtf(vv * (1.f / 96.f) + 1e-5f);

  float y0 = d0 * rs * ln_g[lane]      + ln_b[lane];
  float y1 = d1 * rs * ln_g[lane + 32] + ln_b[lane + 32];
  float y2 = d2 * rs * ln_g[lane + 64] + ln_b[lane + 64];

  // A-fragment scatter: row l, K-chunk kc = ch/32, k_local = lane for all three.
  int hs, j, hh;
  a_map(lane, hs, j, hh);
  int tile  = l >> 4;
  int dlane = (l & 15) + 16 * hs;
  int half  = j * 2 + hh;
  Afrag[((size_t)(tile * 3 + 0) * 32 + dlane) * 16 + half] = (_Float16)y0;
  Afrag[((size_t)(tile * 3 + 1) * 32 + dlane) * 16 + half] = (_Float16)y1;
  Afrag[((size_t)(tile * 3 + 2) * 32 + dlane) * 16 + half] = (_Float16)y2;
}

// ---------------------------------------------------------------------------
// WMMA GEMM, one wave computes a 16 x (16*NPW) strip so each A-fragment load
// feeds NPW v_wmma ops.  All B-fragments for the GEMM are staged once per
// workgroup into LDS (<= 72KB, trivially inside the 320KB WGP LDS) so the
// steady state is: global A loads + ds_load B + v_wmma.
template <int NPW>
__global__ void gemm_wmma(const _Float16* __restrict__ Afrag,
                          const _Float16* __restrict__ Bfrag,
                          float* __restrict__ D,
                          int Mtiles, int Ntiles, int KC, int ldd) {
  extern __shared__ _Float16 Bsh[];
  // Cooperative 16-byte copies of the full B-fragment set into LDS.
  int nvec = (Ntiles * KC * 512) / 8;       // halves / 8 per int4
  for (int i = threadIdx.x; i < nvec; i += blockDim.x)
    ((int4*)Bsh)[i] = ((const int4*)Bfrag)[i];
  __syncthreads();

  int wave = (blockIdx.x * blockDim.x + threadIdx.x) >> 5;
  int lane = threadIdx.x & 31;
  int Ngroups = Ntiles / NPW;
  if (wave >= Mtiles * Ngroups) return;     // wave-uniform: EXEC stays all-ones
  int mt = wave / Ngroups;
  int ng = wave % Ngroups;

  v8f acc[NPW];
#pragma unroll
  for (int u = 0; u < NPW; ++u) acc[u] = (v8f){};

  const v16h* Ap = (const v16h*)Afrag;
  const v16h* Bp = (const v16h*)Bsh;
  for (int kc = 0; kc < KC; ++kc) {
    v16h a = Ap[((size_t)mt * KC + kc) * 32 + lane];
#pragma unroll
    for (int u = 0; u < NPW; ++u) {
      v16h b = Bp[((size_t)(ng * NPW + u) * KC + kc) * 32 + lane];
      acc[u] = __builtin_amdgcn_wmma_f32_16x16x32_f16(
          /*neg_a=*/false, a, /*neg_b=*/false, b,
          /*c_mod=*/(short)0, acc[u], /*reuse_a=*/false, /*reuse_b=*/false);
    }
  }
  // D layout: lane -> col N = nt*16 + lane%16; VGPR r -> row M = r + 8*(lane/16)
  int rbase = (lane >> 4) * 8;
#pragma unroll
  for (int u = 0; u < NPW; ++u) {
    int col = (ng * NPW + u) * 16 + (lane & 15);
#pragma unroll
    for (int r = 0; r < 8; ++r)
      D[(size_t)(mt * 16 + rbase + r) * ldd + col] = acc[u][r];
  }
}

// ---------------------------------------------------------------------------
// Depthwise causal conv (k=4) over the xr half of XZ, fused SiLU.
__global__ void conv_silu(const float* __restrict__ XZ, const float* __restrict__ conv_w,
                          const float* __restrict__ conv_b, float* __restrict__ XC) {
  int idx = blockIdx.x * blockDim.x + threadIdx.x;
  if (idx >= L_SEQ * DINNER) return;
  int l = idx / DINNER, d = idx % DINNER;
  float acc = conv_b[d];
#pragma unroll
  for (int k = 0; k < 4; ++k) {
    int ll = l - 3 + k;
    if (ll >= 0) acc += XZ[(size_t)ll * LDXZ + d] * conv_w[d * 4 + k];
  }
  XC[idx] = acc / (1.f + __expf(-acc));
}

// dt = softplus(dbc[:, :6] @ W_dt^T + b_dt)
__global__ void dt_softplus(const float* __restrict__ DBC, const float* __restrict__ W_dt,
                            const float* __restrict__ b_dt, float* __restrict__ DT) {
  int idx = blockIdx.x * blockDim.x + threadIdx.x;
  if (idx >= L_SEQ * DINNER) return;
  int l = idx / DINNER, d = idx % DINNER;
  float acc = b_dt[d];
#pragma unroll
  for (int r = 0; r < DTRANK; ++r)
    acc += DBC[(size_t)l * LDDBC + r] * W_dt[d * DTRANK + r];
  DT[idx] = (acc > 20.f) ? acc : log1pf(__expf(acc));
}

// ---------------------------------------------------------------------------
// Chunked parallel scan. Recurrence per (d,n): h <- a*h + b,
// a = exp(dt*A), b = dt*u*B.  Chunk composition (P,Q): h_end = P*h_start + Q.
// Phase A: per-(d,n,chunk) compose 128 steps.  786k parallel items.
__global__ void scan_chunks(const float* __restrict__ DT, const float* __restrict__ XC,
                            const float* __restrict__ DBC, const float* __restrict__ Aneg,
                            float* __restrict__ P, float* __restrict__ Q) {
  int t = blockIdx.x * blockDim.x + threadIdx.x;
  if (t >= DINNER * DSTATE * NCHUNK) return;
  int n = t & 15;
  int d = (t >> 4) % DINNER;
  int chunk = t / (DINNER * DSTATE);
  float ac = Aneg[d * DSTATE + n];
  float p = 1.f, q = 0.f;
  int base = chunk * CHUNK;
  for (int i = 0; i < CHUNK; ++i) {
    int l = base + i;
    float dt_t = DT[(size_t)l * DINNER + d];
    float b = dt_t * XC[(size_t)l * DINNER + d] * DBC[(size_t)l * LDDBC + DTRANK + n];
    float a = __expf(dt_t * ac);
    p *= a;
    q = a * q + b;
  }
  P[t] = p;
  Q[t] = q;
}

// Phase B: sequential prefix over 256 chunks, 3072-wide.
__global__ void scan_chunk_prefix(const float* __restrict__ P, const float* __restrict__ Q,
                                  float* __restrict__ H0) {
  int t = blockIdx.x * blockDim.x + threadIdx.x;   // t = d*16 + n
  if (t >= DINNER * DSTATE) return;
  float h = 0.f;
  for (int c = 0; c < NCHUNK; ++c) {
    size_t idx = (size_t)c * (DINNER * DSTATE) + t;
    H0[idx] = h;
    h = P[idx] * h + Q[idx];
  }
}

// Phase C: replay chunk with correct h0, 16 states in registers, fuse D-skip
// and SiLU(z) gate. Writes gated y in place of DT (same (l,d) slot, read-first).
__global__ void scan_final(float* DTY, const float* __restrict__ XC,
                           const float* __restrict__ DBC, const float* __restrict__ Aneg,
                           const float* __restrict__ H0, const float* __restrict__ XZ,
                           const float* __restrict__ D_par) {
  int t = blockIdx.x * blockDim.x + threadIdx.x;
  if (t >= DINNER * NCHUNK) return;
  int d = t % DINNER;
  int chunk = t / DINNER;
  float h[DSTATE], an[DSTATE];
#pragma unroll
  for (int n = 0; n < DSTATE; ++n) {
    h[n]  = H0[(size_t)chunk * (DINNER * DSTATE) + d * DSTATE + n];
    an[n] = Aneg[d * DSTATE + n];
  }
  float Dp = D_par[d];
  int base = chunk * CHUNK;
  for (int i = 0; i < CHUNK; ++i) {
    int l = base + i;
    const float* bc = DBC + (size_t)l * LDDBC;
    __builtin_prefetch(bc + LDDBC, 0, 0);
    float dt_t = DTY[(size_t)l * DINNER + d];
    float u    = XC[(size_t)l * DINNER + d];
    float du = dt_t * u;
    float y = 0.f;
#pragma unroll
    for (int n = 0; n < DSTATE; ++n) {
      float a = __expf(dt_t * an[n]);
      h[n] = h[n] * a + du * bc[DTRANK + n];
      y += h[n] * bc[DTRANK + DSTATE + n];
    }
    y += u * Dp;
    float z = XZ[(size_t)l * LDXZ + DINNER + d];
    y *= z / (1.f + __expf(-z));
    DTY[(size_t)l * DINNER + d] = y;
  }
}

// ---------------------------------------------------------------------------
__global__ void scatter_add(const float* __restrict__ Yout, const int* __restrict__ perm,
                            float* __restrict__ out, int axis) {
  int idx = blockIdx.x * blockDim.x + threadIdx.x;
  if (idx >= L_SEQ * NCH) return;
  int l = idx / NCH, c = idx % NCH;
  int vox = vox_map(perm[l], axis);
  out[(size_t)c * L_SEQ + vox] += Yout[(size_t)l * NCH + c];
}

// ---------------------------------------------------------------------------
extern "C" void kernel_launch(void* const* d_in, const int* in_sizes, int n_in,
                              void* d_out, int out_size, void* d_ws, size_t ws_size,
                              hipStream_t stream) {
  const float* x      = (const float*)d_in[0];
  const float* ln_g   = (const float*)d_in[1];
  const float* ln_b   = (const float*)d_in[2];
  const float* W_in   = (const float*)d_in[3];
  const float* conv_w = (const float*)d_in[4];
  const float* conv_b = (const float*)d_in[5];
  const float* W_x    = (const float*)d_in[6];
  const float* W_dt   = (const float*)d_in[7];
  const float* b_dt   = (const float*)d_in[8];
  const float* A_log  = (const float*)d_in[9];
  const float* D_par  = (const float*)d_in[10];
  const float* W_out  = (const float*)d_in[11];
  float* out = (float*)d_out;

  char* p = (char*)d_ws;
  auto carve = [&](size_t bytes) {
    char* r = p;
    p += (bytes + 255) & ~(size_t)255;
    return (void*)r;
  };
  int*       perm   = (int*)      carve((size_t)L_SEQ * 4);
  float*     Aneg   = (float*)    carve((size_t)DINNER * DSTATE * 4);
  _Float16*  BfWin  = (_Float16*) carve((size_t)24 * 3 * 512 * 2);
  _Float16*  BfWx   = (_Float16*) carve((size_t)3  * 6 * 512 * 2);
  _Float16*  BfWout = (_Float16*) carve((size_t)6  * 6 * 512 * 2);
  _Float16*  Afrag  = (_Float16*) carve((size_t)MTILES * 6 * 512 * 2);
  float*     XZ     = (float*)    carve((size_t)L_SEQ * LDXZ * 4);
  float*     XC     = (float*)    carve((size_t)L_SEQ * DINNER * 4);
  float*     DTY    = (float*)    carve((size_t)L_SEQ * DINNER * 4);
  float*     DBC    = (float*)    carve((size_t)L_SEQ * LDDBC * 4);
  float*     Pb     = (float*)    carve((size_t)DINNER * DSTATE * NCHUNK * 4);
  float*     Qb     = (float*)    carve((size_t)DINNER * DSTATE * NCHUNK * 4);
  float*     H0     = (float*)    carve((size_t)DINNER * DSTATE * NCHUNK * 4);
  float*     OUTA   = (float*)    carve((size_t)L_SEQ * NCH * 4);

  const int TB = 256;
  const size_t shWin  = (size_t)24 * 3 * 512 * 2;  // 73728 B
  const size_t shWx   = (size_t)3  * 6 * 512 * 2;  // 18432 B
  const size_t shWout = (size_t)6  * 6 * 512 * 2;  // 36864 B

  // One-time setup
  build_perm<<<1, 16, 0, stream>>>(perm);
  neg_exp<<<(DINNER * DSTATE + TB - 1) / TB, TB, 0, stream>>>(A_log, Aneg);
  pack_b<<<(24 * 3 * 32 + TB - 1) / TB, TB, 0, stream>>>(W_in,  BfWin,  NCH,    3, 2 * DINNER,          24);
  pack_b<<<(3  * 6 * 32 + TB - 1) / TB, TB, 0, stream>>>(W_x,   BfWx,   DINNER, 6, DTRANK + 2 * DSTATE, 3);
  pack_b<<<(6  * 6 * 32 + TB - 1) / TB, TB, 0, stream>>>(W_out, BfWout, DINNER, 6, NCH,                 6);
  init_out<<<(NCH * L_SEQ + TB - 1) / TB, TB, 0, stream>>>(x, out);

  for (int axis = 0; axis < 3; ++axis) {
    // Gather + LayerNorm, emit A-fragments directly (KC=3)
    gather_ln_pack<<<L_SEQ / 8, TB, 0, stream>>>(x, perm, ln_g, ln_b, Afrag, axis);
    // xz = T @ W_in^T -> [L x 384] : 24 N-tiles, 4 per wave -> 6 groups
    gemm_wmma<4><<<(MTILES * 6) / 8, TB, shWin, stream>>>(Afrag, BfWin, XZ, MTILES, 24, 3, LDXZ);
    // causal depthwise conv + SiLU -> XC [L x 192]
    conv_silu<<<(L_SEQ * DINNER + TB - 1) / TB, TB, 0, stream>>>(XZ, conv_w, conv_b, XC);
    // dbc = XC @ W_x^T -> [L x 48] (cols 38..47 zero-padded): 3 N-tiles, 1 group
    pack_a<<<(MTILES * 6 * 32 + TB - 1) / TB, TB, 0, stream>>>(XC, Afrag, DINNER, 6);
    gemm_wmma<3><<<(MTILES * 1) / 8, TB, shWx, stream>>>(Afrag, BfWx, DBC, MTILES, 3, 6, LDDBC);
    // dt = softplus(dbc[:, :6] @ W_dt^T + b_dt)
    dt_softplus<<<(L_SEQ * DINNER + TB - 1) / TB, TB, 0, stream>>>(DBC, W_dt, b_dt, DTY);
    // chunked parallel selective scan (A: per-chunk compose, B: prefix, C: replay+gate)
    scan_chunks<<<(DINNER * DSTATE * NCHUNK + TB - 1) / TB, TB, 0, stream>>>(DTY, XC, DBC, Aneg, Pb, Qb);
    scan_chunk_prefix<<<(DINNER * DSTATE + TB - 1) / TB, TB, 0, stream>>>(Pb, Qb, H0);
    scan_final<<<(DINNER * NCHUNK + TB - 1) / TB, TB, 0, stream>>>(DTY, XC, DBC, Aneg, H0, XZ, D_par);
    // out_axis = Ygated @ W_out^T -> [L x 96] : 6 N-tiles, 3 per wave -> 2 groups
    pack_a<<<(MTILES * 6 * 32 + TB - 1) / TB, TB, 0, stream>>>(DTY, Afrag, DINNER, 6);
    gemm_wmma<3><<<(MTILES * 2) / 8, TB, shWout, stream>>>(Afrag, BfWout, OUTA, MTILES, 6, 6, NCH);
    // scatter-add back into the volume (bijective per axis -> no atomics)
    scatter_add<<<(L_SEQ * NCH + TB - 1) / TB, TB, 0, stream>>>(OUTA, perm, out, axis);
  }
}
